// NT_Xent_755914244118
// MI455X (gfx1250) — matrix operand order repaired and checked by reading, not compiled, not run on hardware
//
#include <hip/hip_runtime.h>
#include <hip/hip_bf16.h>

#define NB 4096   // batch (B)
#define NN 8192   // 2*B
#define DD 256    // embedding dim
#define SMS 272   // LDS row stride in halves (256 + 16 pad -> conflict-free)
#define WPB 16    // waves per block in the fused kernel (512 threads)

typedef __attribute__((ext_vector_type(16))) _Float16 v16h;
typedef __attribute__((ext_vector_type(8)))  _Float16 v8h;
typedef __attribute__((ext_vector_type(8)))  float    v8f;

// ---------------------------------------------------------------- zero scratch
__global__ void zero_f32(float* __restrict__ p, int n) {
  int i = blockIdx.x * blockDim.x + threadIdx.x;
  if (i < n) p[i] = 0.0f;
}

// ------------------------------------------- normalize + fold temp, fp32->f16
// One wave per row. zn[row] = f16( z[row] * sqrt(2) / max(||z[row]||, eps) )
// so that zn@zn^T directly equals sim/TEMP (TEMP = 0.5).
__global__ __launch_bounds__(256) void normalize_kernel(
    const float* __restrict__ zi, const float* __restrict__ zj,
    _Float16* __restrict__ zn) {
  int wave = threadIdx.x >> 5;
  int lane = threadIdx.x & 31;
  int row  = blockIdx.x * 8 + wave;                       // 0..NN-1
  const float* src = (row < NB) ? (zi + (size_t)row * DD)
                                : (zj + (size_t)(row - NB) * DD);
  float4 a = ((const float4*)src)[lane * 2 + 0];
  float4 b = ((const float4*)src)[lane * 2 + 1];
  float ss = a.x*a.x + a.y*a.y + a.z*a.z + a.w*a.w
           + b.x*b.x + b.y*b.y + b.z*b.z + b.w*b.w;
  #pragma unroll
  for (int off = 1; off < 32; off <<= 1) ss += __shfl_xor(ss, off, 32);
  float scale = 1.41421356237f / fmaxf(sqrtf(ss), 1e-8f); // sqrt(1/TEMP)=sqrt2
  v8h o;
  o[0] = (_Float16)(a.x * scale); o[1] = (_Float16)(a.y * scale);
  o[2] = (_Float16)(a.z * scale); o[3] = (_Float16)(a.w * scale);
  o[4] = (_Float16)(b.x * scale); o[5] = (_Float16)(b.y * scale);
  o[6] = (_Float16)(b.z * scale); o[7] = (_Float16)(b.w * scale);
  *(v8h*)(zn + (size_t)row * DD + lane * 8) = o;
}

// --------------------------- fused sim GEMM (WMMA f16) + online exp-sum + pos
// Block = 16 waves sharing one n-chunk of 64 n-tiles; each wave owns one m-tile.
// B strips (16 rows x 256 halves = 8KB) are double-buffered through LDS so the
// 16 waves read each B tile from L2 exactly once per block (16x traffic cut,
// ~128 MB total L2 traffic for the whole 8192x8192 sim sweep).
__global__ __launch_bounds__(512) void simlse_kernel(
    const _Float16* __restrict__ zn,
    float* __restrict__ rowsum, float* __restrict__ possum) {
  __shared__ __align__(32) _Float16 smem[2][16 * SMS];

  int tid  = threadIdx.x;
  int wave = tid >> 5;
  int lane = tid & 31;
  int h = lane >> 4;                // lane half (K-split selector)
  int r = lane & 15;                // row (A) / column (B,C,D) within tile
  int m0 = (blockIdx.x * WPB + wave) * 16;
  int nt0 = blockIdx.y * 64;        // first n-tile of this chunk

  // Preload all 8 A fragments for this 16-row strip (K = 0..255).
  // A layout (matches B): half h holds K {k0+16h..+15} contiguous (32B).
  const _Float16* arow = zn + (size_t)(m0 + r) * DD;
  v16h Afrag[8];
  #pragma unroll
  for (int kk = 0; kk < 8; ++kk) {
    int k0 = kk * 32;
    Afrag[kk] = *(const v16h*)(arow + k0 + 16 * h);
  }

  // Cooperative staging map: each of 512 threads copies one 16B chunk of the
  // 8KB B strip (16 rows x 512B).
  int crow   = tid >> 5;            // 0..15 (row within strip)
  int cchunk = tid & 31;            // 0..31 (8-half chunk within row)
  const _Float16* gsrc = zn + ((size_t)(nt0 * 16 + crow)) * DD + cchunk * 8;
  _Float16* sdst0 = &smem[0][crow * SMS + cchunk * 8];
  _Float16* sdst1 = &smem[1][crow * SMS + cchunk * 8];

  // Prologue: stage tile 0 into buffer 0.
  *(v8h*)sdst0 = *(const v8h*)gsrc;
  __syncthreads();

  float rs[8], ps[8];
  int   gm[8], pc[8];
  #pragma unroll
  for (int j = 0; j < 8; ++j) {
    rs[j] = 0.0f; ps[j] = 0.0f;
    gm[j] = m0 + 8 * h + j;                 // global row of c[j] in this lane
    pc[j] = (gm[j] + NB) & (NN - 1);        // positive-pair column
  }

  for (int t = 0; t < 64; ++t) {
    int p = t & 1;
    // Prefetch next tile's global chunk into regs (uniform; clamp at the end
    // so EXEC stays all-ones and no OOB read occurs — result is discarded).
    int tn = (t < 63) ? (t + 1) : t;
    v8h pre = *(const v8h*)(gsrc + (size_t)tn * 16 * DD);

    // Pull all 8 B fragments from LDS (ds_load_b128 pairs, pipelined with WMMA).
    // B layout (matches A): half h holds K {k0+16h..+15} contiguous (32B).
    const _Float16* sb = &smem[p][r * SMS];
    v16h Bfrag[8];
    #pragma unroll
    for (int kk = 0; kk < 8; ++kk)
      Bfrag[kk] = *(const v16h*)(sb + kk * 32 + 16 * h);

    v8f c = {};
    #pragma unroll
    for (int kk = 0; kk < 8; ++kk)
      c = __builtin_amdgcn_wmma_f32_16x16x32_f16(
              /*neg_a=*/false, Afrag[kk], /*neg_b=*/false, Bfrag[kk],
              /*c_mod=*/(short)0, c, /*reuse_a=*/false, /*reuse_b=*/false);

    // Commit prefetched strip into the other buffer while WMMAs retire.
    *(v8h*)(p ? sdst0 : sdst1) = pre;

    int gn = (nt0 + t) * 16 + r;
    #pragma unroll
    for (int j = 0; j < 8; ++j) {
      float s = c[j];                              // sim (already /TEMP)
      if (gn != gm[j]) rs[j] += __expf(s - 2.0f);  // masked diag; 2 >= row max
      if (gn == pc[j]) ps[j] += s;                 // one hit per row globally
    }
    __syncthreads();
  }

  // Reduce across the 16 lanes of each half (xor<16 stays within half).
  #pragma unroll
  for (int off = 1; off < 16; off <<= 1) {
    #pragma unroll
    for (int j = 0; j < 8; ++j) {
      rs[j] += __shfl_xor(rs[j], off, 32);
      ps[j] += __shfl_xor(ps[j], off, 32);
    }
  }
  if (r == 0) {
    #pragma unroll
    for (int j = 0; j < 8; ++j) {
      atomicAdd(&rowsum[gm[j]], rs[j]);
      atomicAdd(&possum[gm[j]], ps[j]);
    }
  }
}

// ------------------------------------------------------------- final reduction
// loss = mean_i( [log(rowsum_i) + 2] - pos_i )
__global__ __launch_bounds__(1024) void finalize_kernel(
    const float* __restrict__ rowsum, const float* __restrict__ possum,
    float* __restrict__ out) {
  __shared__ float sdata[1024];
  float acc = 0.0f;
  for (int i = threadIdx.x; i < NN; i += 1024)
    acc += logf(rowsum[i]) + 2.0f - possum[i];
  sdata[threadIdx.x] = acc;
  __syncthreads();
  for (int s = 512; s > 0; s >>= 1) {
    if (threadIdx.x < s) sdata[threadIdx.x] += sdata[threadIdx.x + s];
    __syncthreads();
  }
  if (threadIdx.x == 0) out[0] = sdata[0] / (float)NN;
}

extern "C" void kernel_launch(void* const* d_in, const int* in_sizes, int n_in,
                              void* d_out, int out_size, void* d_ws, size_t ws_size,
                              hipStream_t stream) {
  const float* zi = (const float*)d_in[0];
  const float* zj = (const float*)d_in[1];
  char* ws = (char*)d_ws;
  _Float16* zn   = (_Float16*)ws;                                   // 4 MB
  float* rowsum  = (float*)(ws + (size_t)NN * DD * sizeof(_Float16));
  float* possum  = rowsum + NN;
  float* out     = (float*)d_out;

  zero_f32<<<(2 * NN + 255) / 256, 256, 0, stream>>>(rowsum, 2 * NN);
  normalize_kernel<<<NN / 8, 256, 0, stream>>>(zi, zj, zn);
  simlse_kernel<<<dim3(NN / 16 / WPB, 8), 512, 0, stream>>>(zn, rowsum, possum);
  finalize_kernel<<<1, 1024, 0, stream>>>(rowsum, possum, out);
}
